// PAM_58428735094867
// MI455X (gfx1250) — matrix operand order repaired and checked by reading, not compiled, not run on hardware
//
#include <hip/hip_runtime.h>

typedef __attribute__((ext_vector_type(16))) _Float16 v16h;
typedef __attribute__((ext_vector_type(8)))  float    v8f;

#define N_TOK  8000
#define QTILE  16
#define KTILE  32
#define WAVES  4                    // 4 waves (wave32) per block
#define NQT    (N_TOK / QTILE)      // 500 query tiles
#define NKT    (N_TOK / KTILE)      // 250 key tiles
#define VSTR   36                   // LDS/ws row stride (f16): 72B = 18 banks, pad + 16B-multiple tile
#define VTILE_B (64 * VSTR * 2)     // 4608 bytes = 9 x 512B async chunks
#define KTILE_B (KTILE * 8 * 2)     // 512 bytes  = 1 x 512B async chunk

union AV { v16h h; unsigned u[8]; };

// ---- 16-lane reductions fully fused into DPP VALU ops (1 instr per step) ----
// step sequence: quad_perm xor1, quad_perm xor2, row_half_mirror, row_mirror
// => full 16-lane-row reduction, no LDS traffic, no canonicalize movs.
__device__ __forceinline__ float row_max16(float x) {
  float y;
  asm("v_max_num_f32_dpp %0, %1, %1 quad_perm:[1,0,3,2] row_mask:0xf bank_mask:0xf bound_ctrl:1"
      : "=v"(y) : "v"(x));
  asm("v_max_num_f32_dpp %0, %1, %1 quad_perm:[2,3,0,1] row_mask:0xf bank_mask:0xf bound_ctrl:1"
      : "=v"(x) : "v"(y));
  asm("v_max_num_f32_dpp %0, %1, %1 row_half_mirror row_mask:0xf bank_mask:0xf bound_ctrl:1"
      : "=v"(y) : "v"(x));
  asm("v_max_num_f32_dpp %0, %1, %1 row_mirror row_mask:0xf bank_mask:0xf bound_ctrl:1"
      : "=v"(x) : "v"(y));
  return x;
}
__device__ __forceinline__ float row_sum16(float x) {
  float y;
  asm("v_add_f32_dpp %0, %1, %1 quad_perm:[1,0,3,2] row_mask:0xf bank_mask:0xf bound_ctrl:1"
      : "=v"(y) : "v"(x));
  asm("v_add_f32_dpp %0, %1, %1 quad_perm:[2,3,0,1] row_mask:0xf bank_mask:0xf bound_ctrl:1"
      : "=v"(x) : "v"(y));
  asm("v_add_f32_dpp %0, %1, %1 row_half_mirror row_mask:0xf bank_mask:0xf bound_ctrl:1"
      : "=v"(y) : "v"(x));
  asm("v_add_f32_dpp %0, %1, %1 row_mirror row_mask:0xf bank_mask:0xf bound_ctrl:1"
      : "=v"(x) : "v"(y));
  return x;
}

// ---------------------------------------------------------------------------
// Projection kernel: K = x@Wb, Q = x@Wc (N x 8) row-major f16;
// V = x@Wd written TRANSPOSED PER KEY-TILE: Vt[kt][chan][key], stride VSTR,
// so each attention key-tile is one contiguous 4608B block for async DMA.
// ---------------------------------------------------------------------------
__global__ __launch_bounds__(256) void proj_kernel(
    const float* __restrict__ x,   // [N][64]
    const float* __restrict__ Wb,  // [64][8]
    const float* __restrict__ Wc,  // [64][8]
    const float* __restrict__ Wd,  // [64][64]
    _Float16* __restrict__ Kp,     // [N][8]
    _Float16* __restrict__ Qp,     // [N][8]
    _Float16* __restrict__ Vpt)    // [NKT][64][VSTR]
{
  int t = blockIdx.x * 256 + threadIdx.x;
  if (t >= N_TOK * 64) return;
  int n  = t >> 6;
  int ch = t & 63;
  const float* xr = x + (size_t)n * 64;

  float av = 0.f;
  #pragma unroll 8
  for (int f = 0; f < 64; ++f) av += xr[f] * Wd[f * 64 + ch];
  Vpt[((size_t)(n >> 5) * 64 + ch) * VSTR + (n & 31)] = (_Float16)av;

  if (ch < 8) {
    float ab = 0.f, ac = 0.f;
    #pragma unroll 8
    for (int f = 0; f < 64; ++f) {
      float xv = xr[f];
      ab += xv * Wb[f * 8 + ch];
      ac += xv * Wc[f * 8 + ch];
    }
    Kp[n * 8 + ch] = (_Float16)ab;
    Qp[n * 8 + ch] = (_Float16)ac;
  }
}

// ---------------------------------------------------------------------------
// Fused flash-attention kernel, double-buffered async-to-LDS staging.
// Wave 0 DMAs key-tile kt+1 (K: 1 chunk, V^T: 9 chunks of 512B) with
// GLOBAL_LOAD_ASYNC_TO_LDS_B128 while all waves run WMMAs on tile kt.
// Softmax denominator rides in a 5th output tile via a constant ones-column.
// ---------------------------------------------------------------------------
__global__ __launch_bounds__(WAVES * 32) void attn_kernel(
    const _Float16* __restrict__ Kp,     // [N][8]
    const _Float16* __restrict__ Qp,     // [N][8]
    const _Float16* __restrict__ Vpt,    // [NKT][64][VSTR]
    const float*    __restrict__ x,      // [N][64] (residual)
    const float*    __restrict__ gammap, // [1]
    float*          __restrict__ out)    // [N][64]
{
  __shared__ __align__(16) _Float16 sK[2][KTILE][8];        // double-buffered key tiles
  __shared__ __align__(16) _Float16 sVt[2][64][VSTR];       // double-buffered V^T tiles
  __shared__ __align__(16) _Float16 sOnes[16][VSTR];        // constant ones-column tile
  __shared__ __align__(16) _Float16 sP[WAVES][QTILE][VSTR]; // per-wave P staging

  const int tid  = threadIdx.x;
  const int wave = tid >> 5;
  const int lane = tid & 31;
  const int lh   = lane >> 4;   // lane half
  const int l16  = lane & 15;

  const int qt    = blockIdx.x * WAVES + wave;   // 125*4 == 500 exactly
  const int qbase = qt * QTILE;

  // ---- constant ones tile: column 0 (chan 64) is 1.0, rest 0 ----
  #pragma unroll
  for (int i = 0; i < 4; ++i) {
    int e   = tid * 4 + i;            // 0..511
    int ch  = e >> 5;
    int key = e & 31;
    sOnes[ch][key] = (ch == 0) ? (_Float16)1.0f : (_Float16)0.0f;
  }

  // ---- async DMA issue: 10 x 512B chunks (V^T then K) for one key tile ----
  const unsigned long long vbase = (unsigned long long)Vpt;
  const unsigned long long kbase = (unsigned long long)Kp;
  const unsigned vlds0 = (unsigned)(size_t)&sVt[0][0][0];
  const unsigned vlds1 = (unsigned)(size_t)&sVt[1][0][0];
  const unsigned klds0 = (unsigned)(size_t)&sK[0][0][0];
  const unsigned klds1 = (unsigned)(size_t)&sK[1][0][0];

  auto issue_tile = [&](int kt2, int buf2) {
    unsigned vlds = (buf2 ? vlds1 : vlds0) + (unsigned)lane * 16u;
    unsigned klds = (buf2 ? klds1 : klds0) + (unsigned)lane * 16u;
    unsigned gv   = (unsigned)kt2 * (unsigned)VTILE_B + (unsigned)lane * 16u;
    unsigned gk   = (unsigned)kt2 * (unsigned)KTILE_B + (unsigned)lane * 16u;
    #pragma unroll
    for (int c = 0; c < 9; ++c) {
      asm volatile("global_load_async_to_lds_b128 %0, %1, %2"
                   :: "v"(vlds + (unsigned)c * 512u),
                      "v"(gv   + (unsigned)c * 512u),
                      "s"(vbase)
                   : "memory");
    }
    asm volatile("global_load_async_to_lds_b128 %0, %1, %2"
                 :: "v"(klds), "v"(gk), "s"(kbase) : "memory");
  };

  if (wave == 0) issue_tile(0, 0);   // prologue: stage tile 0

  // ---- Q as WMMA A-matrix (16 x K32, only K=0..7 nonzero) ----
  AV qa;
  #pragma unroll
  for (int j = 0; j < 8; ++j) qa.u[j] = 0u;
  if (lh == 0) {
    const unsigned* qr = (const unsigned*)(Qp + (size_t)(qbase + l16) * 8);
    #pragma unroll
    for (int j = 0; j < 4; ++j) qa.u[j] = qr[j];
  }

  v8f Oa[5] = {};   // Oa[0..3]: ctx accumulator; Oa[4] col 0: softmax denom l
  float m8[8];
  #pragma unroll
  for (int k = 0; k < 8; ++k) m8[k] = -1e30f;

  for (int kt = 0; kt < NKT; ++kt) {
    const int buf = kt & 1;

    // All waves finished reading buf^1 in the previous iteration (kt-1); safe to refill it.
    __syncthreads();
    if (wave == 0) {
      if (kt + 1 < NKT) {
        issue_tile(kt + 1, buf ^ 1);
        asm volatile("s_wait_asynccnt 0xa" ::: "memory");  // tile kt's 10 done (in-order)
      } else {
        asm volatile("s_wait_asynccnt 0x0" ::: "memory");
      }
    }
    __syncthreads();   // publish buf to all waves

    // ---- S = Q * K^T : two 16x16 tiles, K-dim padded from 8 ----
    AV kb0, kb1;
    #pragma unroll
    for (int j = 0; j < 8; ++j) { kb0.u[j] = 0u; kb1.u[j] = 0u; }
    if (lh == 0) {
      const unsigned* k0 = (const unsigned*)&sK[buf][l16][0];
      const unsigned* k1 = (const unsigned*)&sK[buf][16 + l16][0];
      #pragma unroll
      for (int j = 0; j < 4; ++j) { kb0.u[j] = k0[j]; kb1.u[j] = k1[j]; }
    }
    v8f S0 = {}, S1 = {};
    S0 = __builtin_amdgcn_wmma_f32_16x16x32_f16(false, qa.h, false, kb0.h, (short)0, S0, false, false);
    S1 = __builtin_amdgcn_wmma_f32_16x16x32_f16(false, qa.h, false, kb1.h, (short)0, S1, false, false);

    // ---- online softmax: row max via fused DPP, P = exp(S - m_new) ----
    float sc[8];
    #pragma unroll
    for (int k = 0; k < 8; ++k) {
      float t    = row_max16(fmaxf(S0[k], S1[k]));
      float mnew = fmaxf(m8[k], t);
      sc[k] = __expf(m8[k] - mnew);
      m8[k] = mnew;
      S0[k] = __expf(S0[k] - mnew);
      S1[k] = __expf(S1[k] - mnew);
    }
    #pragma unroll
    for (int k = 0; k < 8; ++k) {
      #pragma unroll
      for (int cb = 0; cb < 5; ++cb) Oa[cb][k] *= sc[k];   // includes denom tile
    }

    // ---- P: C-layout -> A-layout via per-wave LDS bounce ----
    #pragma unroll
    for (int k = 0; k < 8; ++k) {
      int row = lh ? (k + 8) : k;
      sP[wave][row][l16]      = (_Float16)S0[k];
      sP[wave][row][16 + l16] = (_Float16)S1[k];
    }
    asm volatile("s_wait_dscnt 0" ::: "memory");  // same-wave DS RAW

    AV pa;
    {
      const unsigned* pr = (const unsigned*)&sP[wave][l16][0];
      #pragma unroll
      for (int j = 0; j < 4; ++j) {
        pa.u[j]     = pr[lh * 4 + j];       // K = lh*8 + 2j, +1
        pa.u[4 + j] = pr[8 + lh * 4 + j];   // K = 16 + lh*8 + 2j, +1
      }
    }

    // ---- O += P * V : 4 ctx tiles + 1 denominator tile, K=32 exact ----
    #pragma unroll
    for (int cb = 0; cb < 5; ++cb) {
      AV vb;
      const unsigned* vr = (cb == 4) ? (const unsigned*)&sOnes[l16][0]
                                     : (const unsigned*)&sVt[buf][cb * 16 + l16][0];
      #pragma unroll
      for (int j = 0; j < 8; ++j) vb.u[j] = vr[lh * 4 + j + (j >= 4 ? 4 : 0)];   // keys lh*8 + 2j (+8 for j>=4), +1
      Oa[cb] = __builtin_amdgcn_wmma_f32_16x16x32_f16(false, pa.h, false, vb.h, (short)0, Oa[cb], false, false);
    }
  }

  // ---- epilogue: l = rowsum(denom tile), out = gamma * O/l + x ----
  const float g = gammap[0];
  #pragma unroll
  for (int k = 0; k < 8; ++k) {
    const float l   = row_sum16(Oa[4][k]);
    const float inv = 1.0f / l;
    const int   row = qbase + (lh ? (k + 8) : k);
    #pragma unroll
    for (int cb = 0; cb < 4; ++cb) {
      const int    col = cb * 16 + l16;
      const size_t idx = (size_t)row * 64 + col;
      out[idx] = g * Oa[cb][k] * inv + x[idx];
    }
  }
}

// ---------------------------------------------------------------------------
extern "C" void kernel_launch(void* const* d_in, const int* in_sizes, int n_in,
                              void* d_out, int out_size, void* d_ws, size_t ws_size,
                              hipStream_t stream) {
  const float* x     = (const float*)d_in[0];
  const float* Wb    = (const float*)d_in[1];
  const float* Wc    = (const float*)d_in[2];
  const float* Wd    = (const float*)d_in[3];
  const float* gamma = (const float*)d_in[4];
  float* out = (float*)d_out;

  _Float16* Kp  = (_Float16*)d_ws;             // N*8
  _Float16* Qp  = Kp + (size_t)N_TOK * 8;      // N*8
  _Float16* Vpt = Qp + (size_t)N_TOK * 8;      // NKT*64*VSTR  (~1.15 MB)

  proj_kernel<<<(N_TOK * 64 + 255) / 256, 256, 0, stream>>>(x, Wb, Wc, Wd, Kp, Qp, Vpt);
  attn_kernel<<<NQT / WAVES, WAVES * 32, 0, stream>>>(Kp, Qp, Vpt, x, gamma, out);
}
